// ConvLSTMBlock_69870527971932
// MI455X (gfx1250) — matrix operand order repaired and checked by reading, not compiled, not run on hardware
//
#include <hip/hip_runtime.h>
#include <hip/hip_bf16.h>
#include <math.h>

// ---------------------------------------------------------------------------
// Problem constants (from reference): B=4, T=16, H=W=64, C=32, F=64, K=3
// Gates G = 4F = 256.  Input-conv GEMM: M = B*T*H*W = 262144, N=256, K=288.
// Step GEMM: M = B*H*W = 16384, N=256, K=576.
// ---------------------------------------------------------------------------
#define BB   4
#define TT   16
#define HH   64
#define WW   64
#define CC   32
#define FF   64
#define GG   256
#define HWP  (HH * WW)            // 4096
#define NPIX_X (BB * TT * HWP)    // 262144  (time folded into batch)
#define NPIX_S (BB * HWP)         // 16384

typedef __attribute__((ext_vector_type(16))) _Float16 v16h;
typedef __attribute__((ext_vector_type(8)))  _Float16 v8h;
typedef __attribute__((ext_vector_type(8)))  float    v8f;

__device__ __forceinline__ float hsig(float x) {
    return fminf(fmaxf(0.2f * x + 0.5f, 0.0f), 1.0f);
}

// Load one 16x32 f16 A fragment for implicit 3x3 conv GEMM.
// Lane layout (ISA 7.12.2): lanes 0-15 -> M=lane, K = {0..7, 16..23};
// lanes 16-31 -> M=lane-16, K = {8..15, 24..31}.  Both per-lane 8-half
// groups fall in the same 3x3 tap because Cin % 32 == 0 here.
// k0 is a compile-time constant after full unroll, so tap/dy/dx fold away.
template <int Cin>
__device__ __forceinline__ v16h load_a_frag(const _Float16* __restrict__ src,
                                            int p, int py, int px,
                                            int k0, int khalf) {
    int kg  = k0 + khalf;
    int tap = kg / Cin;
    int c   = kg - tap * Cin;
    int tq  = tap / 3;
    int dy  = tq - 1, dx = tap - 3 * tq - 1;
    int ny  = py + dy, nx = px + dx;
    v16h a = {};
    if (ny >= 0 && ny < HH && nx >= 0 && nx < WW) {
        const _Float16* ap = src + (size_t)(p + dy * WW + dx) * Cin + c;
        union { v16h v; v8h h[2]; } U;
        U.h[0] = *(const v8h*)(ap);
        U.h[1] = *(const v8h*)(ap + 16);
        a = U.v;
    }
    return a;
}

// ---------------------------------------------------------------------------
// Weight pack: [k][n] f32  ->  [n][k] f16   (k = tap*Cin + c, tap = 3*dy+dx)
// ---------------------------------------------------------------------------
__global__ __launch_bounds__(256)
void packw_kernel(const float* __restrict__ src, _Float16* __restrict__ dst, int Ktot) {
    int idx = blockIdx.x * 256 + threadIdx.x;
    int total = Ktot * GG;
    if (idx >= total) return;
    int n = idx / Ktot;
    int k = idx - n * Ktot;
    dst[idx] = (_Float16)src[(size_t)k * GG + n];
}

// ---------------------------------------------------------------------------
// LayerNorm over C=32 channels, f32 -> normalized f16 (WMMA A operand feed)
// ---------------------------------------------------------------------------
__global__ __launch_bounds__(256)
void ln_kernel(const float* __restrict__ x, const float* __restrict__ gamma,
               const float* __restrict__ beta, _Float16* __restrict__ xn) {
    int p = blockIdx.x * 256 + threadIdx.x;   // one pixel per thread
    const float* xp = x + (size_t)p * CC;
    float v[CC];
    float s = 0.f, s2 = 0.f;
#pragma unroll
    for (int i = 0; i < CC / 4; ++i) {
        float4 q = ((const float4*)xp)[i];
        v[4*i+0] = q.x; v[4*i+1] = q.y; v[4*i+2] = q.z; v[4*i+3] = q.w;
        s  += q.x + q.y + q.z + q.w;
        s2 += q.x*q.x + q.y*q.y + q.z*q.z + q.w*q.w;
    }
    float mean = s * (1.0f / CC);
    float var  = s2 * (1.0f / CC) - mean * mean;
    float rs   = rsqrtf(var + 1e-3f);
    _Float16* op = xn + (size_t)p * CC;
#pragma unroll
    for (int c = 0; c < CC; ++c)
        op[c] = (_Float16)((v[c] - mean) * rs * gamma[c] + beta[c]);
}

// ---------------------------------------------------------------------------
// Input conv as implicit GEMM:  zx[p][n] = sum_k xn_patch[p][k] * WxT[n][k] + b[n]
// 4 waves / WG.  WG tile = 32(M) x 256(N).  Each wave: 2 M-tiles x 4 N-tiles
// = 8 accumulators; each B fragment reused twice in registers; fully unrolled
// K-chain (9 steps) covers the f16-WMMA hazard window with real work.
// ---------------------------------------------------------------------------
__global__ __launch_bounds__(128)
void convx_kernel(const _Float16* __restrict__ xn, const _Float16* __restrict__ wT,
                  const float* __restrict__ bias, _Float16* __restrict__ zx) {
    constexpr int Cin = CC, Ktot = 9 * CC;    // 288
    const int wg_m0 = blockIdx.x * 32;
    const int wave  = threadIdx.x >> 5;       // 0..3 -> 64-col gate group
    const int lane  = threadIdx.x & 31;
    const int n_base = wave * 64;

    const int p0   = wg_m0 + (lane & 15);     // M-tile 0 row
    const int p1   = p0 + 16;                 // M-tile 1 row
    const int rem0 = p0 & (HWP - 1), rem1 = p1 & (HWP - 1);
    const int py0  = rem0 >> 6, px0 = rem0 & (WW - 1);
    const int py1  = rem1 >> 6, px1 = rem1 & (WW - 1);

    const int khalf = (lane < 16) ? 0 : 8;    // A half-select
    const int kq    = (lane < 16) ? 0 : 8;    // B half-select
    const _Float16* wbase = wT + (size_t)(n_base + (lane & 15)) * Ktot + kq;

    v8f acc[2][4] = {};

#pragma unroll
    for (int k0 = 0; k0 < Ktot; k0 += 32) {
        v16h a0 = load_a_frag<Cin>(xn, p0, py0, px0, k0, khalf);
        v16h a1 = load_a_frag<Cin>(xn, p1, py1, px1, k0, khalf);
#pragma unroll
        for (int nt = 0; nt < 4; ++nt) {
            const _Float16* bp = wbase + (size_t)nt * 16 * Ktot + k0;
            union { v16h v; v8h h[2]; } UB;
            UB.h[0] = *(const v8h*)(bp);
            UB.h[1] = *(const v8h*)(bp + 16);
            v16h bf = UB.v;
            acc[0][nt] = __builtin_amdgcn_wmma_f32_16x16x32_f16(
                false, a0, false, bf, (short)0, acc[0][nt], false, false);
            acc[1][nt] = __builtin_amdgcn_wmma_f32_16x16x32_f16(
                false, a1, false, bf, (short)0, acc[1][nt], false, false);
        }
    }

    // Epilogue: D[r] -> row (r + 8*(lane>=16)), col (lane&15); add bias.
    const int rowAdd = (lane >> 4) * 8;
    const int col0   = lane & 15;
#pragma unroll
    for (int ms = 0; ms < 2; ++ms) {
#pragma unroll
        for (int nt = 0; nt < 4; ++nt) {
            int n = n_base + nt * 16 + col0;
            float bs = bias[n];
#pragma unroll
            for (int r = 0; r < 8; ++r) {
                int pp = wg_m0 + ms * 16 + r + rowAdd;
                zx[(size_t)pp * GG + n] = (_Float16)(acc[ms][nt][r] + bs);
            }
        }
    }
}

// ---------------------------------------------------------------------------
// One LSTM step: z = zx[t] + conv(h_prev, Wh) via WMMA; gates fused in LDS.
// h double-buffered (hin/hout) to avoid cross-block RAW on the conv halo.
// ---------------------------------------------------------------------------
__global__ __launch_bounds__(128)
void step_kernel(const _Float16* __restrict__ hin, const _Float16* __restrict__ wT,
                 const _Float16* __restrict__ zx, float* __restrict__ cbuf,
                 _Float16* __restrict__ hout, float* __restrict__ hfull, int t) {
    constexpr int Cin = FF, Ktot = 9 * FF;    // 576
    __shared__ float zlds[32 * GG];           // 32 KB

    const int wg_m0 = blockIdx.x * 32;        // q base, q in [0, B*H*W)
    const int wave  = threadIdx.x >> 5;
    const int lane  = threadIdx.x & 31;
    const int n_base = wave * 64;

    const int q0   = wg_m0 + (lane & 15);
    const int q1   = q0 + 16;
    const int rem0 = q0 & (HWP - 1), rem1 = q1 & (HWP - 1);
    const int py0  = rem0 >> 6, px0 = rem0 & (WW - 1);
    const int py1  = rem1 >> 6, px1 = rem1 & (WW - 1);

    const int khalf = (lane < 16) ? 0 : 8;
    const int kq    = (lane < 16) ? 0 : 8;
    const _Float16* wbase = wT + (size_t)(n_base + (lane & 15)) * Ktot + kq;

    v8f acc[2][4] = {};

#pragma unroll
    for (int k0 = 0; k0 < Ktot; k0 += 32) {
        v16h a0 = load_a_frag<Cin>(hin, q0, py0, px0, k0, khalf);
        v16h a1 = load_a_frag<Cin>(hin, q1, py1, px1, k0, khalf);
#pragma unroll
        for (int nt = 0; nt < 4; ++nt) {
            const _Float16* bp = wbase + (size_t)nt * 16 * Ktot + k0;
            union { v16h v; v8h h[2]; } UB;
            UB.h[0] = *(const v8h*)(bp);
            UB.h[1] = *(const v8h*)(bp + 16);
            v16h bf = UB.v;
            acc[0][nt] = __builtin_amdgcn_wmma_f32_16x16x32_f16(
                false, a0, false, bf, (short)0, acc[0][nt], false, false);
            acc[1][nt] = __builtin_amdgcn_wmma_f32_16x16x32_f16(
                false, a1, false, bf, (short)0, acc[1][nt], false, false);
        }
    }

    // Epilogue 1: z = acc + zx  ->  LDS (all 4 gates of a pixel in one WG)
    const int rowAdd = (lane >> 4) * 8;
    const int col0   = lane & 15;
#pragma unroll
    for (int ms = 0; ms < 2; ++ms) {
#pragma unroll
        for (int nt = 0; nt < 4; ++nt) {
            int col = n_base + nt * 16 + col0;
#pragma unroll
            for (int r = 0; r < 8; ++r) {
                int mloc = ms * 16 + r + rowAdd;
                int qq   = wg_m0 + mloc;
                int b2   = qq >> 12;              // / HWP
                int r2   = qq & (HWP - 1);
                int gp   = ((b2 << 4) + t) * HWP + r2;   // pixel in [B,T,H,W]
                float z  = acc[ms][nt][r] + (float)zx[(size_t)gp * GG + col];
                zlds[mloc * GG + col] = z;
            }
        }
    }
    __syncthreads();

    // Epilogue 2: elementwise LSTM update for 32 pixels x 64 features
#pragma unroll
    for (int e = 0; e < 16; ++e) {
        int idx  = threadIdx.x + 128 * e;     // 0..2047
        int mloc = idx >> 6;
        int f    = idx & 63;
        int qq   = wg_m0 + mloc;
        float zi = zlds[mloc * GG + f];
        float zf = zlds[mloc * GG + 64 + f];
        float zc = zlds[mloc * GG + 128 + f];
        float zo = zlds[mloc * GG + 192 + f];
        size_t sidx = (size_t)qq * FF + f;
        float cold = cbuf[sidx];
        float ig = hsig(zi);
        float fg = hsig(zf);
        float cn = fg * cold + ig * tanhf(zc);
        float og = hsig(zo);
        float h  = og * tanhf(cn);
        cbuf[sidx]  = cn;
        hfull[sidx] = h;
        hout[sidx]  = (_Float16)h;
    }
}

// ---------------------------------------------------------------------------
// 2x2 spatial max-pool of h(t) -> d_out[b][t][y][x][f]
// ---------------------------------------------------------------------------
__global__ __launch_bounds__(256)
void pool_kernel(const float* __restrict__ hfull, float* __restrict__ out, int t) {
    int idx = blockIdx.x * 256 + threadIdx.x;         // B*32*32*64 = 262144
    int f = idx & 63;
    int x = (idx >> 6) & 31;
    int y = (idx >> 11) & 31;
    int b = idx >> 16;
    int q00 = b * HWP + (2 * y) * WW + 2 * x;
    float m = hfull[(size_t)q00 * FF + f];
    m = fmaxf(m, hfull[(size_t)(q00 + 1)      * FF + f]);
    m = fmaxf(m, hfull[(size_t)(q00 + WW)     * FF + f]);
    m = fmaxf(m, hfull[(size_t)(q00 + WW + 1) * FF + f]);
    out[(((size_t)(b * TT + t)) * 1024 + y * 32 + x) * FF + f] = m;
}

// ---------------------------------------------------------------------------
extern "C" void kernel_launch(void* const* d_in, const int* in_sizes, int n_in,
                              void* d_out, int out_size, void* d_ws, size_t ws_size,
                              hipStream_t stream) {
    const float* x     = (const float*)d_in[0];
    const float* gamma = (const float*)d_in[1];
    const float* beta  = (const float*)d_in[2];
    const float* Wx    = (const float*)d_in[3];
    const float* Wh    = (const float*)d_in[4];
    const float* bias  = (const float*)d_in[5];
    float* out = (float*)d_out;
    char*  ws  = (char*)d_ws;

    size_t off = 0;
    auto take = [&](size_t bytes) -> char* {
        char* p = ws + off;
        off += (bytes + 255) & ~(size_t)255;
        return p;
    };
    _Float16* wxT   = (_Float16*)take((size_t)GG * 288 * 2);       // 144 KB
    _Float16* whT   = (_Float16*)take((size_t)GG * 576 * 2);       // 288 KB
    _Float16* xn    = (_Float16*)take((size_t)NPIX_X * CC * 2);    // 16.8 MB
    _Float16* zx    = (_Float16*)take((size_t)NPIX_X * GG * 2);    // 134 MB
    _Float16* hA    = (_Float16*)take((size_t)NPIX_S * FF * 2);    // 2.1 MB
    _Float16* hB    = (_Float16*)take((size_t)NPIX_S * FF * 2);    // 2.1 MB
    float*    cbuf  = (float*)   take((size_t)NPIX_S * FF * 4);    // 4.2 MB
    float*    hfull = (float*)   take((size_t)NPIX_S * FF * 4);    // 4.2 MB

    // Zero-init recurrent state every call (graph-capture-safe memset nodes).
    hipMemsetAsync(hA,   0, (size_t)NPIX_S * FF * 2, stream);
    hipMemsetAsync(hB,   0, (size_t)NPIX_S * FF * 2, stream);
    hipMemsetAsync(cbuf, 0, (size_t)NPIX_S * FF * 4, stream);

    packw_kernel<<<(288 * GG + 255) / 256, 256, 0, stream>>>(Wx, wxT, 288);
    packw_kernel<<<(576 * GG + 255) / 256, 256, 0, stream>>>(Wh, whT, 576);
    ln_kernel<<<NPIX_X / 256, 256, 0, stream>>>(x, gamma, beta, xn);
    convx_kernel<<<NPIX_X / 32, 128, 0, stream>>>(xn, wxT, bias, zx);

    for (int t = 0; t < TT; ++t) {
        const _Float16* hin = (t & 1) ? hB : hA;
        _Float16* hout      = (t & 1) ? hA : hB;
        step_kernel<<<NPIX_S / 32, 128, 0, stream>>>(hin, whT, zx, cbuf, hout, hfull, t);
        pool_kernel<<<(BB * 32 * 32 * FF) / 256, 256, 0, stream>>>(hfull, out, t);
    }
}
